// HGNN_24816321036989
// MI455X (gfx1250) — matrix operand order repaired and checked by reading, not compiled, not run on hardware
//
#include <hip/hip_runtime.h>
#include <math.h>

typedef __attribute__((ext_vector_type(16))) _Float16 v16h;
typedef __attribute__((ext_vector_type(8)))  _Float16 v8h;
typedef __attribute__((ext_vector_type(8)))  float    v8f;

#define N_A   20000
#define N_APAD 20032      // pad to multiple of 64 rows
#define E_A   40000       // multiple of 64
#define N_F   8000        // multiple of 64
#define E_F   16000       // multiple of 64
#define N_B   256
#define IN_DIM 295
#define IN_PAD 320
#define EPSV 1e-5f

// ---------------------------------------------------------------------------
// WMMA fragment helpers (wave32; C/D per CDNA5 ISA 7.12.2, A/B in this code's contiguous k order k = half*16 + i)
//
// A (16x32 f16, MxK): lane = half*16 + m holds row m; per-lane data is two
// contiguous 8-half runs: k in [half*16, half*16+8) and [half*16+8, half*16+16)
// -> exactly 2 x b128 loads.
//
// B (32x16 f16, KxN): lane (n = lane&15, half = lane>>4), element i holds
// B[half*16+i][n]. We pre-swizzle weights in global memory to
// [frag][lane][i] so this is one contiguous 32-byte read per lane.
//
// C/D f32 16x16: VGPR v: lanes 0-15 -> M=v, N=lane; lanes 16-31 -> M=v+8.
// ---------------------------------------------------------------------------
__device__ inline v16h load_a16(const _Float16* row /* &A[m][k0] */, int half) {
    v8h lo = *(const v8h*)(row + half * 16);
    v8h hi = *(const v8h*)(row + half * 16 + 8);
    return __builtin_shufflevector(lo, hi, 0, 1, 2, 3, 4, 5, 6, 7,
                                   8, 9, 10, 11, 12, 13, 14, 15);
}

__device__ inline float sigf(float x) { return 1.f / (1.f + expf(-x)); }

// ---------------------------------------------------------------------------
// Generic WMMA GEMM: C[M,N] = act(A[M,K] @ B[K,N] + bias)
// A row-major f16 (M%64==0, K%32==0). Bf is FRAGMENT-PACKED weights:
//   Bf[((kc*(N/16) + ct)*32 + lane)*16 + i] = B[kc*32 + (lane>>4)*16 + i][ct*16 + (lane&15)]
// Block = 128 threads (4 waves); each wave one 16-row x 64-col tile.
// grid = (M/64, N/64).
// ---------------------------------------------------------------------------
__global__ void k_gemm(const _Float16* __restrict__ A, const _Float16* __restrict__ Bf,
                       const float* __restrict__ bias, float* __restrict__ C,
                       _Float16* __restrict__ C16, int M, int N, int K, int act) {
    __shared__ _Float16 As[64 * 32];
    int tid = threadIdx.x, wave = tid >> 5, lane = tid & 31;
    int row0 = blockIdx.x * 64;
    int ct0 = blockIdx.y * 4;           // first 16-col tile of this block
    int ntiles = N >> 4;
    int half = lane >> 4;
    v8f acc[4] = {};
    for (int k0 = 0; k0 < K; k0 += 32) {
        // stage A 64x32 block: 16-byte vector loads, fully coalesced
#pragma unroll
        for (int t = 0; t < 2; ++t) {
            int idx = t * 1024 + tid * 8;
            int r = idx >> 5, c = idx & 31;
            *(v8h*)(As + idx) = *(const v8h*)(A + (size_t)(row0 + r) * K + k0 + c);
        }
        __syncthreads();
        v16h a = load_a16(&As[(wave * 16 + (lane & 15)) * 32], half);
        const _Float16* fb = Bf + (((size_t)(k0 >> 5) * ntiles + ct0) * 32 + lane) * 16;
        if (k0 + 32 < K)
            __builtin_prefetch(fb + (size_t)ntiles * 512, 0, 3);
#pragma unroll
        for (int nt = 0; nt < 4; ++nt) {
            v16h b = *(const v16h*)(fb + nt * 512);   // 32B contiguous -> 2 x b128
            acc[nt] = __builtin_amdgcn_wmma_f32_16x16x32_f16(
                false, a, false, b, (short)0, acc[nt], false, false);
        }
        __syncthreads();
    }
    int nlo = lane & 15, mhi = half * 8;
#pragma unroll
    for (int nt = 0; nt < 4; ++nt) {
#pragma unroll
        for (int v = 0; v < 8; ++v) {
            int m = row0 + wave * 16 + mhi + v;
            int n = (ct0 + nt) * 16 + nlo;
            float val = acc[nt][v] + (bias ? bias[n] : 0.f);
            if (act) val = fmaxf(val, 0.f);
            C[(size_t)m * N + n] = val;
            if (C16) C16[(size_t)m * N + n] = (_Float16)val;
        }
    }
}

// ---------------------------------------------------------------------------
// Fused NNConv message kernel.
//   msg[e,i] = sum_c coef[e,c] * ( Hsrc[e,:] @ Wbig[:, c*64+i] )
// Wf is the fragment-packed Wbig [64 x C*64] (same packing as k_gemm's Bf,
// ntiles = C*4). Per 16-edge wave: A fragments live in registers for the
// whole kernel; the C channel-blocks of Wf stream straight from global
// (they stay hot in WGP$/L2 across all waves); coef-weighted accumulation
// is fused into the WMMA epilogue so the [E, C*64] intermediate never
// touches memory. No LDS on the B path, no syncthreads in the main loop.
// ---------------------------------------------------------------------------
__global__ void k_edge_mpnn(const _Float16* __restrict__ Hsrc /*[Epad,64]*/,
                            const _Float16* __restrict__ Wf,
                            const float* __restrict__ coef /*[Epad,16]*/,
                            float* __restrict__ msg /*[Epad,64]*/, int C) {
    __shared__ float Lc[4 * 256];        // per-wave coef tile, layout [c][row]
    int tid = threadIdx.x, wave = tid >> 5, lane = tid & 31;
    int e0 = blockIdx.x * 64 + wave * 16;
    for (int t = lane; t < 256; t += 32)
        Lc[wave * 256 + (t & 15) * 16 + (t >> 4)] =
            coef[(size_t)(e0 + (t >> 4)) * 16 + (t & 15)];
    int m = lane & 15, half = lane >> 4;
    const _Float16* rowp = Hsrc + (size_t)(e0 + m) * 64;
    v16h a0 = load_a16(rowp, half);
    v16h a1 = load_a16(rowp + 32, half);
    __syncthreads();
    v8f acc[4] = {};
    int nlo = lane & 15, mhi = half * 8;
    int ntiles = C * 4;
    for (int c = 0; c < C; ++c) {
        const _Float16* f0 = Wf + ((size_t)(c * 4) * 32 + lane) * 16;            // kc=0
        const _Float16* f1 = Wf + (((size_t)ntiles + c * 4) * 32 + lane) * 16;   // kc=1
        float cf[8];
#pragma unroll
        for (int v = 0; v < 8; ++v) cf[v] = Lc[wave * 256 + c * 16 + mhi + v];
#pragma unroll
        for (int nt = 0; nt < 4; ++nt) {
            v16h b0 = *(const v16h*)(f0 + nt * 512);
            v16h b1 = *(const v16h*)(f1 + nt * 512);
            v8f t = {};
            t = __builtin_amdgcn_wmma_f32_16x16x32_f16(false, a0, false, b0,
                                                       (short)0, t, false, false);
            t = __builtin_amdgcn_wmma_f32_16x16x32_f16(false, a1, false, b1,
                                                       (short)0, t, false, false);
#pragma unroll
            for (int v = 0; v < 8; ++v)
                acc[nt][v] += cf[v] * t[v];
        }
    }
#pragma unroll
    for (int nt = 0; nt < 4; ++nt)
#pragma unroll
        for (int v = 0; v < 8; ++v)
            msg[(size_t)(e0 + mhi + v) * 64 + nt * 16 + nlo] = acc[nt][v];
}

// ---------------------------------------------------------------------------
// Weight packing into WMMA-fragment order (global memory)
// ---------------------------------------------------------------------------
// Generic: W f32 [Kin, N] row-major -> f16 fragments for [Kpad, N]
__global__ void k_pack_w(const float* __restrict__ W, _Float16* __restrict__ out,
                         int Kin, int Kpad, int N) {
    int total = Kpad * N;
    int idx = blockIdx.x * blockDim.x + threadIdx.x;
    if (idx >= total) return;
    int i = idx & 15;
    int lane = (idx >> 4) & 31;
    int frag = idx >> 9;
    int ntiles = N >> 4;
    int ct = frag % ntiles, kc = frag / ntiles;
    int k = kc * 32 + (lane >> 4) * 16 + i;
    int n = ct * 16 + (lane & 15);
    out[idx] = (_Float16)((k < Kin) ? W[(size_t)k * N + n] : 0.f);
}
// NNConv: Wbig[j][c*64+io] = Wnet[c][io*64+j] (c<Cin) | bnet[io*64+j] (c==Cin)
__global__ void k_pack_wnet(const float* __restrict__ Wnet, const float* __restrict__ bnet,
                            _Float16* __restrict__ out, int Cin) {
    int C = Cin + 1;
    int total = 64 * C * 64;
    int idx = blockIdx.x * blockDim.x + threadIdx.x;
    if (idx >= total) return;
    int i = idx & 15;
    int lane = (idx >> 4) & 31;
    int frag = idx >> 9;
    int ntiles = C * 4;
    int ct = frag % ntiles, kc = frag / ntiles;
    int k = kc * 32 + (lane >> 4) * 16 + i;          // contraction dim j
    int col = ct * 16 + (lane & 15);
    int c = col >> 6, io = col & 63;
    float v = (c < Cin) ? Wnet[(size_t)c * 4096 + io * 64 + k] : bnet[io * 64 + k];
    out[idx] = (_Float16)v;
}

// ---------------------------------------------------------------------------
// Small scalar kernels
// ---------------------------------------------------------------------------
__global__ void k_setf(float* p, float v, int n) {
    int i = blockIdx.x * blockDim.x + threadIdx.x;
    if (i < n) p[i] = v;
}
// pad/convert f32 [rows,cols] -> f16 [rowsPad,colsPad] (zero pad) — activations
__global__ void k_pad_convert(const float* __restrict__ X, _Float16* __restrict__ Y,
                              int rows, int cols, int rowsPad, int colsPad) {
    int idx = blockIdx.x * blockDim.x + threadIdx.x;
    if (idx >= rowsPad * colsPad) return;
    int r = idx / colsPad, c = idx % colsPad;
    float v = (r < rows && c < cols) ? X[(size_t)r * cols + c] : 0.f;
    Y[idx] = (_Float16)v;
}
__global__ void k_build_coef(const float* __restrict__ ef, int Cin,
                             float* __restrict__ coef, int E) {
    int idx = blockIdx.x * blockDim.x + threadIdx.x;
    if (idx >= E * 16) return;
    int e = idx >> 4, c = idx & 15;
    float v = 0.f;
    if (c < Cin) v = ef[(size_t)e * Cin + c];
    else if (c == Cin) v = 1.f;
    coef[idx] = v;
}
__global__ void k_gather_f16(const float* __restrict__ H, const int* __restrict__ src,
                             _Float16* __restrict__ out, int E) {
    int idx = blockIdx.x * blockDim.x + threadIdx.x;
    if (idx >= E * 64) return;
    int e = idx >> 6, d = idx & 63;
    out[idx] = (_Float16)H[(size_t)src[e] * 64 + d];
}
__global__ void k_scatter_add(const float* __restrict__ msg, const int* __restrict__ dst,
                              float* __restrict__ out, int E) {
    int idx = blockIdx.x * blockDim.x + threadIdx.x;
    if (idx >= E * 64) return;
    int e = idx >> 6, d = idx & 63;
    atomicAdd(&out[(size_t)dst[e] * 64 + d], msg[idx]);
}
__global__ void k_colstats(const float* __restrict__ X, int M, int N,
                           float* __restrict__ mean, float* __restrict__ var) {
    __shared__ float ss[256], sq[256];
    int col = blockIdx.x;
    float s = 0.f, s2 = 0.f;
    for (int r = threadIdx.x; r < M; r += 256) {
        float v = X[(size_t)r * N + col];
        s += v; s2 += v * v;
    }
    ss[threadIdx.x] = s; sq[threadIdx.x] = s2;
    __syncthreads();
    for (int st = 128; st > 0; st >>= 1) {
        if (threadIdx.x < st) { ss[threadIdx.x] += ss[threadIdx.x + st];
                                sq[threadIdx.x] += sq[threadIdx.x + st]; }
        __syncthreads();
    }
    if (threadIdx.x == 0) {
        float m = ss[0] / (float)M;
        mean[col] = m;
        var[col] = sq[0] / (float)M - m * m;   // population var (ddof=0)
    }
}
__global__ void k_bn_relu(const float* __restrict__ X, const float* __restrict__ mean,
                          const float* __restrict__ var, const float* __restrict__ g,
                          const float* __restrict__ be, float* __restrict__ Y,
                          _Float16* __restrict__ Y16, int M, int Mpad) {
    int idx = blockIdx.x * blockDim.x + threadIdx.x;
    if (idx >= Mpad * 64) return;
    int m = idx >> 6, c = idx & 63;
    float o = 0.f;
    if (m < M) {
        float v = X[idx];
        o = fmaxf(g[c] * (v - mean[c]) * rsqrtf(var[c] + EPSV) + be[c], 0.f);
    }
    if (Y) Y[idx] = o;
    if (Y16) Y16[idx] = (_Float16)o;
}
// 2-slot multi-head attention combine (HEADS=4, DK=16, scores /16)
__global__ void k_attn(const float* __restrict__ q, const float* __restrict__ k1,
                       const float* __restrict__ k2, const float* __restrict__ v1,
                       const float* __restrict__ v2, float* __restrict__ out,
                       _Float16* __restrict__ out16, int N) {
    int idx = blockIdx.x * blockDim.x + threadIdx.x;
    if (idx >= N * 4) return;
    int node = idx >> 2, h = idx & 3;
    size_t off = (size_t)node * 64 + h * 16;
    float s1 = 0.f, s2 = 0.f;
    for (int d = 0; d < 16; ++d) {
        float qv = q[off + d];
        s1 += qv * k1[off + d];
        s2 += qv * k2[off + d];
    }
    s1 *= (1.f / 16.f); s2 *= (1.f / 16.f);
    float mx = fmaxf(s1, s2);
    float e1 = expf(s1 - mx), e2 = expf(s2 - mx);
    float inv = 1.f / (e1 + e2);
    float a1 = e1 * inv, a2 = e2 * inv;
    for (int d = 0; d < 16; ++d) {
        float o = a1 * v1[off + d] + a2 * v2[off + d];
        out[off + d] = o;
        out16[off + d] = (_Float16)o;
    }
}
// GRU cell, torch gate order (r,z,n); in-place update of h (and f16 mirror)
__global__ void k_gru_gate(const float* __restrict__ gi, const float* __restrict__ gh,
                           float* __restrict__ h, _Float16* __restrict__ h16,
                           int Nvalid, int Npad) {
    int idx = blockIdx.x * blockDim.x + threadIdx.x;
    if (idx >= Npad * 64) return;
    int m = idx >> 6, d = idx & 63;
    float out = 0.f;
    if (m < Nvalid) {
        const float* gim = gi + (size_t)m * 192;
        const float* ghm = gh + (size_t)m * 192;
        float r = sigf(gim[d] + ghm[d]);
        float z = sigf(gim[64 + d] + ghm[64 + d]);
        float n = tanhf(gim[128 + d] + r * ghm[128 + d]);
        out = (1.f - z) * n + z * h[idx];
    }
    h[idx] = out;
    if (h16) h16[idx] = (_Float16)out;
}
// out[m*stride+off] = X[m,:K] . w + b[0]
__global__ void k_rowdot(const float* __restrict__ X, const float* __restrict__ w,
                         const float* __restrict__ b, float* __restrict__ out,
                         int M, int K, int stride, int off) {
    int m = blockIdx.x * blockDim.x + threadIdx.x;
    if (m >= M) return;
    float s = 0.f;
    for (int k = 0; k < K; ++k) s += X[(size_t)m * K + k] * w[k];
    out[(size_t)m * stride + off] = s + b[0];
}
__device__ inline void atomicMaxF(float* addr, float val) {
    int* ia = (int*)addr;
    int cur = __float_as_int(*addr);
    while (__int_as_float(cur) < val) {
        int prev = atomicCAS(ia, cur, __float_as_int(val));
        if (prev == cur) break;
        cur = prev;
    }
}
__global__ void k_seg_max(const float* __restrict__ z, const int* __restrict__ gid,
                          float* __restrict__ zmax, int N) {
    int i = blockIdx.x * blockDim.x + threadIdx.x;
    if (i < N) atomicMaxF(&zmax[gid[i]], z[i]);
}
__global__ void k_seg_expsum(float* __restrict__ z, const float* __restrict__ zmax,
                             const int* __restrict__ gid, float* __restrict__ den, int N) {
    int i = blockIdx.x * blockDim.x + threadIdx.x;
    if (i >= N) return;
    float e = expf(z[i] - zmax[gid[i]]);
    z[i] = e;                                   // reuse z as e
    atomicAdd(&den[gid[i]], e);
}
// att_pool (cols 0..63) and sum_pool (cols 64..127) into rep[B,295]
__global__ void k_pool(const float* __restrict__ x, const float* __restrict__ e,
                       const float* __restrict__ den, const int* __restrict__ gid,
                       float* __restrict__ rep, int N) {
    int idx = blockIdx.x * blockDim.x + threadIdx.x;
    if (idx >= N * 64) return;
    int n = idx >> 6, d = idx & 63;
    int g = gid[n];
    float xv = x[idx];
    float alpha = e[n] / den[g];
    atomicAdd(&rep[(size_t)g * IN_DIM + d], alpha * xv);
    atomicAdd(&rep[(size_t)g * IN_DIM + 64 + d], xv);
}
__global__ void k_copy_mf(const float* __restrict__ mf, float* __restrict__ rep, int B) {
    int idx = blockIdx.x * blockDim.x + threadIdx.x;
    if (idx >= B * 167) return;
    int b = idx / 167, c = idx % 167;
    rep[(size_t)b * IN_DIM + 128 + c] = mf[idx];
}
// w = sigmoid((joint @ W1 + b1) @ W2 + b2), joint[B,2], W1[2,64], W2[64,2]
__global__ void k_sc(const float* __restrict__ joint, const float* __restrict__ W1,
                     const float* __restrict__ b1, const float* __restrict__ W2,
                     const float* __restrict__ b2, float* __restrict__ wsc, int B) {
    int b = blockIdx.x * blockDim.x + threadIdx.x;
    if (b >= B) return;
    float j0 = joint[b * 2], j1 = joint[b * 2 + 1];
    float o0 = b2[0], o1 = b2[1];
    for (int k = 0; k < 64; ++k) {
        float h = j0 * W1[k] + j1 * W1[64 + k] + b1[k];
        o0 += h * W2[k * 2];
        o1 += h * W2[k * 2 + 1];
    }
    wsc[b * 2]     = sigf(o0);
    wsc[b * 2 + 1] = sigf(o1);
}
__global__ void k_scale_rep(float* __restrict__ rep, const float* __restrict__ wsc, int slot) {
    int idx = blockIdx.x * blockDim.x + threadIdx.x;
    if (idx >= N_B * IN_DIM) return;
    rep[idx] *= wsc[(idx / IN_DIM) * 2 + slot];
}

// ---------------------------------------------------------------------------
extern "C" void kernel_launch(void* const* d_in, const int* in_sizes, int n_in,
                              void* d_out, int out_size, void* d_ws, size_t ws_size,
                              hipStream_t stream) {
    (void)in_sizes; (void)n_in; (void)out_size; (void)ws_size;
    const float* af  = (const float*)d_in[0];
    const float* bf  = (const float*)d_in[1];
    const float* fnf = (const float*)d_in[2];
    const float* fef = (const float*)d_in[3];
    const float* mf  = (const float*)d_in[4];
    const int* atom_src   = (const int*)d_in[5];
    const int* atom_dst   = (const int*)d_in[6];
    const int* fg_src     = (const int*)d_in[7];
    const int* fg_dst     = (const int*)d_in[8];
    const int* a2f_dst    = (const int*)d_in[9];
    const int* atom_graph = (const int*)d_in[10];
    const int* fg_graph   = (const int*)d_in[11];
    enum {
        enc_a_W1, enc_a_b1, enc_a_g1, enc_a_be1, enc_a_W2, enc_a_b2, enc_a_g2, enc_a_be2,
        enc_f_W1, enc_f_b1, enc_f_g1, enc_f_be1, enc_f_W2, enc_f_b2, enc_f_g2, enc_f_be2,
        at_Wnet, at_bnet, fg_Wnet, fg_bnet,
        gru_a_Wih, gru_a_bih, gru_a_Whh, gru_a_bhh,
        gru_f_Wih, gru_f_bih, gru_f_Whh, gru_f_bhh,
        la_Wq, la_Wk, la_Wv, la_Wo, la_bo,
        ro_a_w, ro_a_b, ro_f_w, ro_f_b,
        sc_a_w, sc_a_b, sc_m_w, sc_m_b, sc_W1, sc_b1, sc_W2, sc_b2,
        out_a_W1, out_a_b1, out_a_W2, out_a_b2,
        out_f_W1, out_f_b1, out_f_W2, out_f_b2, NPARAM
    };
    const float* P[NPARAM];
    for (int i = 0; i < NPARAM; ++i) P[i] = (const float*)d_in[12 + i];

    // ---- workspace bump allocator --------------------------------------
    char* base = (char*)d_ws;
    size_t off = 0;
    auto alloc  = [&](size_t bytes) { void* p = base + off;
                                      off += (bytes + 255) & ~(size_t)255; return p; };
    auto F32 = [&](size_t n) { return (float*)alloc(n * 4); };
    auto F16 = [&](size_t n) { return (_Float16*)alloc(n * 2); };

    _Float16 *af16 = F16((size_t)N_APAD * 64), *fnf16 = F16((size_t)N_F * 96);
    _Float16 *w1a16 = F16(64 * 64), *w2a16 = F16(64 * 64);
    _Float16 *w1f16 = F16(96 * 64), *w2f16 = F16(64 * 64);
    _Float16 *wbigA = F16(64 * 960), *wbigF = F16(64 * 448);
    _Float16 *wq16 = F16(64 * 64), *wk16 = F16(64 * 64), *wv16 = F16(64 * 64), *wo16 = F16(64 * 64);
    _Float16 *gAih = F16(64 * 192), *gAhh = F16(64 * 192), *gFih = F16(64 * 192), *gFhh = F16(64 * 192);
    _Float16 *oW1a = F16(IN_PAD * 256), *oW1f = F16(IN_PAD * 256);
    float *tmpA = F32((size_t)N_APAD * 64), *tmpF = F32((size_t)N_F * 64);
    float *meanb = F32(64), *varb = F32(64);
    _Float16 *h1a16 = F16((size_t)N_APAD * 64), *h1f16 = F16((size_t)N_F * 64);
    float *uaf = F32((size_t)N_APAD * 64), *ufnf = F32((size_t)N_F * 64);
    _Float16 *uaf16 = F16((size_t)N_APAD * 64), *ufnf16 = F16((size_t)N_F * 64);
    _Float16 *hsA16 = F16((size_t)E_A * 64), *hsF16 = F16((size_t)E_F * 64);
    float *coefA = F32((size_t)E_A * 16), *coefF = F32((size_t)E_F * 16);
    float *msgA = F32((size_t)E_A * 64), *msgF = F32((size_t)E_F * 64);
    float *uam = F32((size_t)N_APAD * 64), *ufnm = F32((size_t)N_F * 64), *aug = F32((size_t)N_F * 64);
    _Float16 *uam16 = F16((size_t)N_APAD * 64), *ufnm16 = F16((size_t)N_F * 64), *aug16 = F16((size_t)N_F * 64);
    float *qb = F32((size_t)N_F * 64), *k1b = F32((size_t)N_F * 64), *k2b = F32((size_t)N_F * 64);
    float *v1b = F32((size_t)N_F * 64), *v2b = F32((size_t)N_F * 64);
    float *attn = F32((size_t)N_F * 64), *lao = F32((size_t)N_F * 64);
    _Float16 *attn16 = F16((size_t)N_F * 64), *lao16 = F16((size_t)N_F * 64);
    float *giA = F32((size_t)N_APAD * 192), *ghA = F32((size_t)N_APAD * 192);
    float *giF = F32((size_t)N_F * 192), *ghF = F32((size_t)N_F * 192);
    float *zA = F32(N_APAD), *zF = F32(N_F);
    float *zmaxA = F32(N_B), *zmaxF = F32(N_B), *denA = F32(N_B), *denF = F32(N_B);
    float *repA = F32((size_t)N_B * IN_DIM), *repF = F32((size_t)N_B * IN_DIM);
    _Float16 *rep16A = F16((size_t)N_B * IN_PAD), *rep16F = F16((size_t)N_B * IN_PAD);
    float *joint = F32(N_B * 2), *wsc = F32(N_B * 2);
    float *hidA = F32((size_t)N_B * 256), *hidF = F32((size_t)N_B * 256);
    float* dout = (float*)d_out;

    auto GRD = [](size_t n) { return dim3((unsigned)((n + 255) / 256)); };
    auto gemm = [&](const _Float16* A, const _Float16* Bw, const float* bias,
                    float* C, _Float16* C16, int M, int N, int K, int act) {
        k_gemm<<<dim3(M / 64, N / 64), dim3(128), 0, stream>>>(A, Bw, bias, C, C16, M, N, K, act);
    };
    auto enc_layer = [&](const _Float16* X16, const _Float16* W16, const float* bias,
                         const float* g, const float* be, float* tmp,
                         float* Y, _Float16* Y16, int M, int Mpad, int K) {
        gemm(X16, W16, bias, tmp, nullptr, Mpad, 64, K, 0);
        k_colstats<<<dim3(64), dim3(256), 0, stream>>>(tmp, M, 64, meanb, varb);
        k_bn_relu<<<GRD((size_t)Mpad * 64), dim3(256), 0, stream>>>(tmp, meanb, varb, g, be, Y, Y16, M, Mpad);
    };

    // ---- weight / input prep (fragment-order f16 repacks) ---------------
    k_pad_convert<<<GRD((size_t)N_APAD * 64), 256, 0, stream>>>(af, af16, N_A, 42, N_APAD, 64);
    k_pad_convert<<<GRD((size_t)N_F * 96), 256, 0, stream>>>(fnf, fnf16, N_F, 82, N_F, 96);
    k_pack_w<<<GRD(64 * 64), 256, 0, stream>>>(P[enc_a_W1], w1a16, 42, 64, 64);
    k_pack_w<<<GRD(64 * 64), 256, 0, stream>>>(P[enc_a_W2], w2a16, 64, 64, 64);
    k_pack_w<<<GRD(96 * 64), 256, 0, stream>>>(P[enc_f_W1], w1f16, 82, 96, 64);
    k_pack_w<<<GRD(64 * 64), 256, 0, stream>>>(P[enc_f_W2], w2f16, 64, 64, 64);
    k_pack_wnet<<<GRD(64 * 960), 256, 0, stream>>>(P[at_Wnet], P[at_bnet], wbigA, 14);
    k_pack_wnet<<<GRD(64 * 448), 256, 0, stream>>>(P[fg_Wnet], P[fg_bnet], wbigF, 6);
    k_pack_w<<<GRD(64 * 64), 256, 0, stream>>>(P[la_Wq], wq16, 64, 64, 64);
    k_pack_w<<<GRD(64 * 64), 256, 0, stream>>>(P[la_Wk], wk16, 64, 64, 64);
    k_pack_w<<<GRD(64 * 64), 256, 0, stream>>>(P[la_Wv], wv16, 64, 64, 64);
    k_pack_w<<<GRD(64 * 64), 256, 0, stream>>>(P[la_Wo], wo16, 64, 64, 64);
    k_pack_w<<<GRD(64 * 192), 256, 0, stream>>>(P[gru_a_Wih], gAih, 64, 64, 192);
    k_pack_w<<<GRD(64 * 192), 256, 0, stream>>>(P[gru_a_Whh], gAhh, 64, 64, 192);
    k_pack_w<<<GRD(64 * 192), 256, 0, stream>>>(P[gru_f_Wih], gFih, 64, 64, 192);
    k_pack_w<<<GRD(64 * 192), 256, 0, stream>>>(P[gru_f_Whh], gFhh, 64, 64, 192);
    k_pack_w<<<GRD((size_t)IN_PAD * 256), 256, 0, stream>>>(P[out_a_W1], oW1a, IN_DIM, IN_PAD, 256);
    k_pack_w<<<GRD((size_t)IN_PAD * 256), 256, 0, stream>>>(P[out_f_W1], oW1f, IN_DIM, IN_PAD, 256);
    k_build_coef<<<GRD((size_t)E_A * 16), 256, 0, stream>>>(bf, 14, coefA, E_A);
    k_build_coef<<<GRD((size_t)E_F * 16), 256, 0, stream>>>(fef, 6, coefF, E_F);

    // ---- encoders -------------------------------------------------------
    enc_layer(af16, w1a16, P[enc_a_b1], P[enc_a_g1], P[enc_a_be1], tmpA, nullptr, h1a16, N_A, N_APAD, 64);
    enc_layer(h1a16, w2a16, P[enc_a_b2], P[enc_a_g2], P[enc_a_be2], tmpA, uaf, uaf16, N_A, N_APAD, 64);
    enc_layer(fnf16, w1f16, P[enc_f_b1], P[enc_f_g1], P[enc_f_be1], tmpF, nullptr, h1f16, N_F, N_F, 96);
    enc_layer(h1f16, w2f16, P[enc_f_b2], P[enc_f_g2], P[enc_f_be2], tmpF, ufnf, ufnf16, N_F, N_F, 64);

    // ---- 2 message-passing steps ---------------------------------------
    for (int step = 0; step < 2; ++step) {
        k_gather_f16<<<GRD((size_t)E_F * 64), 256, 0, stream>>>(ufnf, fg_src, hsF16, E_F);
        k_gather_f16<<<GRD((size_t)E_A * 64), 256, 0, stream>>>(uaf, atom_src, hsA16, E_A);
        k_edge_mpnn<<<dim3(E_F / 64), dim3(128), 0, stream>>>(hsF16, wbigF, coefF, msgF, 7);
        k_edge_mpnn<<<dim3(E_A / 64), dim3(128), 0, stream>>>(hsA16, wbigA, coefA, msgA, 15);
        k_setf<<<GRD((size_t)N_F * 64), 256, 0, stream>>>(ufnm, 0.f, N_F * 64);
        k_scatter_add<<<GRD((size_t)E_F * 64), 256, 0, stream>>>(msgF, fg_dst, ufnm, E_F);
        k_setf<<<GRD((size_t)N_APAD * 64), 256, 0, stream>>>(uam, 0.f, N_APAD * 64);
        k_scatter_add<<<GRD((size_t)E_A * 64), 256, 0, stream>>>(msgA, atom_dst, uam, E_A);
        k_setf<<<GRD((size_t)N_F * 64), 256, 0, stream>>>(aug, 0.f, N_F * 64);
        k_scatter_add<<<GRD((size_t)N_A * 64), 256, 0, stream>>>(uam, a2f_dst, aug, N_A);
        k_pad_convert<<<GRD((size_t)N_F * 64), 256, 0, stream>>>(ufnm, ufnm16, N_F, 64, N_F, 64);
        k_pad_convert<<<GRD((size_t)N_APAD * 64), 256, 0, stream>>>(uam, uam16, N_A, 64, N_APAD, 64);
        k_pad_convert<<<GRD((size_t)N_F * 64), 256, 0, stream>>>(aug, aug16, N_F, 64, N_F, 64);
        // local augmentation: q from motif state; kv = [aug (fine), ufnm (coarse)]
        gemm(ufnf16, wq16, nullptr, qb, nullptr, N_F, 64, 64, 0);
        gemm(aug16,  wk16, nullptr, k1b, nullptr, N_F, 64, 64, 0);
        gemm(ufnm16, wk16, nullptr, k2b, nullptr, N_F, 64, 64, 0);
        gemm(aug16,  wv16, nullptr, v1b, nullptr, N_F, 64, 64, 0);
        gemm(ufnm16, wv16, nullptr, v2b, nullptr, N_F, 64, 64, 0);
        k_attn<<<GRD((size_t)N_F * 4), 256, 0, stream>>>(qb, k1b, k2b, v1b, v2b, attn, attn16, N_F);
        gemm(attn16, wo16, P[la_bo], lao, lao16, N_F, 64, 64, 0);
        // GRUs (motif uses augmented message; atoms use raw message)
        gemm(lao16,  gFih, P[gru_f_bih], giF, nullptr, N_F, 192, 64, 0);
        gemm(ufnf16, gFhh, P[gru_f_bhh], ghF, nullptr, N_F, 192, 64, 0);
        k_gru_gate<<<GRD((size_t)N_F * 64), 256, 0, stream>>>(giF, ghF, ufnf, ufnf16, N_F, N_F);
        gemm(uam16, gAih, P[gru_a_bih], giA, nullptr, N_APAD, 192, 64, 0);
        gemm(uaf16, gAhh, P[gru_a_bhh], ghA, nullptr, N_APAD, 192, 64, 0);
        k_gru_gate<<<GRD((size_t)N_APAD * 64), 256, 0, stream>>>(giA, ghA, uaf, uaf16, N_A, N_APAD);
    }

    // ---- readouts -------------------------------------------------------
    k_rowdot<<<GRD(N_A), 256, 0, stream>>>(uaf, P[ro_a_w], P[ro_a_b], zA, N_A, 64, 1, 0);
    k_rowdot<<<GRD(N_F), 256, 0, stream>>>(ufnf, P[ro_f_w], P[ro_f_b], zF, N_F, 64, 1, 0);
    k_setf<<<GRD(N_B), 256, 0, stream>>>(zmaxA, -3.0e38f, N_B);
    k_setf<<<GRD(N_B), 256, 0, stream>>>(zmaxF, -3.0e38f, N_B);
    k_seg_max<<<GRD(N_A), 256, 0, stream>>>(zA, atom_graph, zmaxA, N_A);
    k_seg_max<<<GRD(N_F), 256, 0, stream>>>(zF, fg_graph, zmaxF, N_F);
    k_setf<<<GRD(N_B), 256, 0, stream>>>(denA, 0.f, N_B);
    k_setf<<<GRD(N_B), 256, 0, stream>>>(denF, 0.f, N_B);
    k_seg_expsum<<<GRD(N_A), 256, 0, stream>>>(zA, zmaxA, atom_graph, denA, N_A);
    k_seg_expsum<<<GRD(N_F), 256, 0, stream>>>(zF, zmaxF, fg_graph, denF, N_F);
    k_setf<<<GRD((size_t)N_B * IN_DIM), 256, 0, stream>>>(repA, 0.f, N_B * IN_DIM);
    k_setf<<<GRD((size_t)N_B * IN_DIM), 256, 0, stream>>>(repF, 0.f, N_B * IN_DIM);
    k_pool<<<GRD((size_t)N_A * 64), 256, 0, stream>>>(uaf, zA, denA, atom_graph, repA, N_A);
    k_pool<<<GRD((size_t)N_F * 64), 256, 0, stream>>>(ufnf, zF, denF, fg_graph, repF, N_F);
    k_copy_mf<<<GRD((size_t)N_B * 167), 256, 0, stream>>>(mf, repA, N_B);
    k_copy_mf<<<GRD((size_t)N_B * 167), 256, 0, stream>>>(mf, repF, N_B);
    // contextual rescale
    k_rowdot<<<GRD(N_B), 256, 0, stream>>>(repA, P[sc_a_w], P[sc_a_b], joint, N_B, IN_DIM, 2, 0);
    k_rowdot<<<GRD(N_B), 256, 0, stream>>>(repF, P[sc_m_w], P[sc_m_b], joint, N_B, IN_DIM, 2, 1);
    k_sc<<<GRD(N_B), 256, 0, stream>>>(joint, P[sc_W1], P[sc_b1], P[sc_W2], P[sc_b2], wsc, N_B);
    k_scale_rep<<<GRD((size_t)N_B * IN_DIM), 256, 0, stream>>>(repA, wsc, 0);
    k_scale_rep<<<GRD((size_t)N_B * IN_DIM), 256, 0, stream>>>(repF, wsc, 1);
    // output MLP heads
    k_pad_convert<<<GRD((size_t)N_B * IN_PAD), 256, 0, stream>>>(repA, rep16A, N_B, IN_DIM, N_B, IN_PAD);
    k_pad_convert<<<GRD((size_t)N_B * IN_PAD), 256, 0, stream>>>(repF, rep16F, N_B, IN_DIM, N_B, IN_PAD);
    gemm(rep16A, oW1a, P[out_a_b1], hidA, nullptr, N_B, 256, IN_PAD, 1);
    gemm(rep16F, oW1f, P[out_f_b1], hidF, nullptr, N_B, 256, IN_PAD, 1);
    k_rowdot<<<GRD(N_B), 256, 0, stream>>>(hidA, P[out_a_W2], P[out_a_b2], dout, N_B, 256, 1, 0);
    k_rowdot<<<GRD(N_B), 256, 0, stream>>>(hidF, P[out_f_W2], P[out_f_b2], dout + N_B, N_B, 256, 1, 0);
}
